// GATEncoder_9921374454406
// MI455X (gfx1250) — compile-verified, matrix operand order checked by reading, not run on hardware
//
#include <hip/hip_runtime.h>
#include <hip/hip_fp16.h>

typedef __attribute__((ext_vector_type(16))) _Float16 v16h;
typedef __attribute__((ext_vector_type(8)))  float    v8f;

#define N_NODES 50000
#define E_REAL  800000
#define E_TOT   (E_REAL + N_NODES)   // +self loops
#define IN_CH   128
#define HID     128
#define OUT_CH  64
#define NEG_SLOPE 0.2f

// ---------------------------------------------------------------- utilities

__global__ void fill_f32(float* __restrict__ p, float v, long n) {
  long i = (long)blockIdx.x * blockDim.x + threadIdx.x;
  long stride = (long)gridDim.x * blockDim.x;
  for (; i < n; i += stride) p[i] = v;
}

__device__ __forceinline__ void atomicMaxF(float* addr, float v) {
  // sign-split float max: both branches monotonically move the stored bits
  // toward the larger float; init must be -inf (0xFF800000).
  if (v >= 0.0f) atomicMax((int*)addr, __float_as_int(v));
  else           atomicMin((unsigned int*)addr, __float_as_uint(v));
}

// ------------------------------------------------- WMMA GEMM + fused attention
// C[MxN] = A[MxK] * B[KxN] (fp32 memory, fp16 WMMA compute), and fused
//   os[m] = C[m,:].a_s ; od[m] = C[m,:].a_d  (attention logit dots)
// One wave computes a full 16xN strip. B is staged once per block into LDS,
// f16-converted and pre-swizzled into WMMA B-fragment order so each wave
// fetches a fragment with two ds_load_b128.
// Fragment layouts per ISA 7.12.2 (16-bit A 16x32): lane<16 holds row M=lane,
// halves {kb..kb+7, kb+16..kb+23}; lane>=16 the +8 K offsets. B mirrors with
// column N in place of M. C/D: vgpr r = row (r + 8*(lane>>4)), col = lane&15.
template<int NT, int K>   // NT = N/16 column tiles; K multiple of 32
__global__ void wmma_gemm_attn(const float* __restrict__ A,
                               const float* __restrict__ B,
                               float* __restrict__ C,
                               const float* __restrict__ a_s,
                               const float* __restrict__ a_d,
                               float* __restrict__ os, float* __restrict__ od,
                               int M) {
  constexpr int N  = NT * 16;
  constexpr int KC = K / 32;
  __shared__ __align__(32) _Float16 ldsB[K * N];   // fragment-ordered

  const int lane   = threadIdx.x & 31;
  const int wlocal = threadIdx.x >> 5;             // 0..7
  const int mtile  = blockIdx.x * 8 + wlocal;
  const int hh  = lane >> 4;
  const int l16 = lane & 15;

  // ---- stage B -> LDS (slot = (kc*NT + nt)*32 + lane; 16 halves per slot)
  for (int slot = threadIdx.x; slot < KC * NT * 32; slot += blockDim.x) {
    const int sl   = slot & 31;
    const int t    = slot >> 5;
    const int nt   = t % NT;
    const int kc   = t / NT;
    const int kb   = kc * 32 + (sl >> 4) * 8;
    const int n    = nt * 16 + (sl & 15);
    _Float16* dst  = ldsB + slot * 16;
#pragma unroll
    for (int j = 0; j < 8; ++j) {
      dst[j]     = (_Float16)B[(size_t)(kb + j) * N + n];
      dst[j + 8] = (_Float16)B[(size_t)(kb + 16 + j) * N + n];
    }
  }
  __syncthreads();
  if (mtile * 16 >= M) return;     // uniform per wave: EXEC stays all-1s

  const float* __restrict__ Arow = A + (size_t)(mtile * 16 + l16) * K;
  v8f acc[NT] = {};
#pragma unroll
  for (int kc = 0; kc < KC; ++kc) {
    const int kb = kc * 32 + hh * 8;
    const float4 a0 = *(const float4*)(Arow + kb);
    const float4 a1 = *(const float4*)(Arow + kb + 4);
    const float4 a2 = *(const float4*)(Arow + kb + 16);
    const float4 a3 = *(const float4*)(Arow + kb + 20);
    const float av[16] = {a0.x, a0.y, a0.z, a0.w, a1.x, a1.y, a1.z, a1.w,
                          a2.x, a2.y, a2.z, a2.w, a3.x, a3.y, a3.z, a3.w};
    v16h a;
#pragma unroll
    for (int j = 0; j < 16; ++j) a[j] = (_Float16)av[j];
#pragma unroll
    for (int nt = 0; nt < NT; ++nt) {
      const v16h b = *(const v16h*)(ldsB + ((size_t)(kc * NT + nt) * 32 + lane) * 16);
      acc[nt] = __builtin_amdgcn_wmma_f32_16x16x32_f16(false, a, false, b,
                                                       (short)0, acc[nt], false, false);
    }
  }

  // ---- epilogue: store C strip + fused attention dots
  float asv[NT], adv[NT];
#pragma unroll
  for (int nt = 0; nt < NT; ++nt) {
    asv[nt] = a_s[nt * 16 + l16];
    adv[nt] = a_d[nt * 16 + l16];
  }
#pragma unroll
  for (int r = 0; r < 8; ++r) {
    const int row = mtile * 16 + hh * 8 + r;
    float* Crow = C + (size_t)row * N + l16;
    float ps = 0.0f, pd = 0.0f;
#pragma unroll
    for (int nt = 0; nt < NT; ++nt) {
      const float v = acc[nt][r];
      Crow[nt * 16] = v;
      ps = fmaf(v, asv[nt], ps);
      pd = fmaf(v, adv[nt], pd);
    }
#pragma unroll
    for (int off = 8; off > 0; off >>= 1) {   // reduce within 16-lane half
      ps += __shfl_xor(ps, off, 32);
      pd += __shfl_xor(pd, off, 32);
    }
    if (l16 == 0) { os[row] = ps; od[row] = pd; }
  }
}

// --------------------------------------------------------------- edge passes
__device__ __forceinline__ void edge_src_dst(const int* __restrict__ ei, int e,
                                             int& s, int& d) {
  if (e < E_REAL) { s = ei[e]; d = ei[E_REAL + e]; }
  else            { s = d = e - E_REAL; }            // virtual self-loop
}

// pass A: e = leaky_relu(asrc[s]+adst[d]); segment max into m[d].
__global__ void edge_logits_max(const int* __restrict__ ei,
    const float* __restrict__ as1, const float* __restrict__ ad1,
    float* __restrict__ eb1, float* __restrict__ m1,
    const float* __restrict__ as2, const float* __restrict__ ad2,
    float* __restrict__ eb2, float* __restrict__ m2) {
  const int e = blockIdx.x * blockDim.x + threadIdx.x;
  if (e >= E_TOT) return;
  int s, d; edge_src_dst(ei, e, s, d);
  float t = as1[s] + ad1[d];
  t = t > 0.0f ? t : NEG_SLOPE * t;
  eb1[e] = t;
  atomicMaxF(m1 + d, t);
  if (as2) {
    float u = as2[s] + ad2[d];
    u = u > 0.0f ? u : NEG_SLOPE * u;
    eb2[e] = u;
    atomicMaxF(m2 + d, u);
  }
}

// pass B: p = exp(e - m[d]); segment sum into den[d]; eb overwritten with p.
__global__ void edge_exp_sum(const int* __restrict__ ei,
    float* __restrict__ eb1, const float* __restrict__ m1, float* __restrict__ dn1,
    float* __restrict__ eb2, const float* __restrict__ m2, float* __restrict__ dn2) {
  const int e = blockIdx.x * blockDim.x + threadIdx.x;
  if (e >= E_TOT) return;
  int s, d; edge_src_dst(ei, e, s, d);
  (void)s;
  float p = __expf(eb1[e] - m1[d]);
  eb1[e] = p;
  atomicAdd(dn1 + d, p);
  if (eb2) {
    float q = __expf(eb2[e] - m2[d]);
    eb2[e] = q;
    atomicAdd(dn2 + d, q);
  }
}

// pass C (single, C=128): out[d] += (p/den[d]) * H[s]
// one wave32 per edge; one float4 gather per lane + 4 L2 fp32 atomics.
__global__ void edge_aggregate128(const int* __restrict__ ei,
    const float* __restrict__ eb, const float* __restrict__ dn,
    const float* __restrict__ H, float* __restrict__ out) {
  const int wave = blockIdx.x * (blockDim.x >> 5) + (threadIdx.x >> 5);
  const int lane = threadIdx.x & 31;
  if (wave >= E_TOT) return;
  int s, d; edge_src_dst(ei, wave, s, d);
  const float coef = eb[wave] / dn[d];
  const float4 h4 = *(const float4*)(H + (size_t)s * 128 + lane * 4);
  float* od = out + (size_t)d * 128 + lane * 4;
  atomicAdd(od + 0, coef * h4.x);
  atomicAdd(od + 1, coef * h4.y);
  atomicAdd(od + 2, coef * h4.z);
  atomicAdd(od + 3, coef * h4.w);
}

// pass C (dual, C=64 each): mu and logstd share edges -> fused.
__global__ void edge_aggregate_dual64(const int* __restrict__ ei,
    const float* __restrict__ eb1, const float* __restrict__ dn1,
    const float* __restrict__ H1, float* __restrict__ out1,
    const float* __restrict__ eb2, const float* __restrict__ dn2,
    const float* __restrict__ H2, float* __restrict__ out2) {
  const int wave = blockIdx.x * (blockDim.x >> 5) + (threadIdx.x >> 5);
  const int lane = threadIdx.x & 31;
  if (wave >= E_TOT) return;
  int s, d; edge_src_dst(ei, wave, s, d);
  {
    const float c1 = eb1[wave] / dn1[d];
    const float2 h2 = *(const float2*)(H1 + (size_t)s * 64 + lane * 2);
    float* od = out1 + (size_t)d * 64 + lane * 2;
    atomicAdd(od + 0, c1 * h2.x);
    atomicAdd(od + 1, c1 * h2.y);
  }
  {
    const float c2 = eb2[wave] / dn2[d];
    const float2 h2 = *(const float2*)(H2 + (size_t)s * 64 + lane * 2);
    float* od = out2 + (size_t)d * 64 + lane * 2;
    atomicAdd(od + 0, c2 * h2.x);
    atomicAdd(od + 1, c2 * h2.y);
  }
}

// h = (relu?) (h + b[c])
__global__ void bias_act(float* __restrict__ h, const float* __restrict__ b,
                         long n, int C, int relu) {
  long i = (long)blockIdx.x * blockDim.x + threadIdx.x;
  if (i >= n) return;
  float v = h[i] + b[(int)(i & (C - 1))];   // C is a power of two (128 / 64)
  if (relu) v = v > 0.0f ? v : 0.0f;
  h[i] = v;
}

// ---------------------------------------------------------------- launch

static inline int ceildiv(long a, long b) { return (int)((a + b - 1) / b); }

extern "C" void kernel_launch(void* const* d_in, const int* in_sizes, int n_in,
                              void* d_out, int out_size, void* d_ws, size_t ws_size,
                              hipStream_t stream) {
  (void)in_sizes; (void)n_in; (void)out_size; (void)ws_size;
  const float* x       = (const float*)d_in[0];
  const int*   ei      = (const int*)d_in[1];     // [2, E] flat: src row, dst row
  const float* W1      = (const float*)d_in[2];
  const float* a_src1  = (const float*)d_in[3];
  const float* a_dst1  = (const float*)d_in[4];
  const float* b1      = (const float*)d_in[5];
  const float* W_mu    = (const float*)d_in[6];
  const float* a_srcm  = (const float*)d_in[7];
  const float* a_dstm  = (const float*)d_in[8];
  const float* b_mu    = (const float*)d_in[9];
  const float* W_ls    = (const float*)d_in[10];
  const float* a_srcl  = (const float*)d_in[11];
  const float* a_dstl  = (const float*)d_in[12];
  const float* b_ls    = (const float*)d_in[13];

  const long SZ_H = (long)N_NODES * HID;      // 6.4M floats
  const long SZ_O = (long)N_NODES * OUT_CH;   // 3.2M floats

  float* ws  = (float*)d_ws;
  float* hx  = ws;                 // [N,128] layer-1 features (pre-aggregate)
  float* h1  = hx + SZ_H;          // [N,128] layer-1 output
  float* hmu = hx;                 // reuse hx after layer-1 aggregation
  float* hls = hx + SZ_O;
  float* sc  = h1 + SZ_H;          // scalar region
  float* as1 = sc +  0L * N_NODES, *ad1 = sc + 1L * N_NODES;
  float* m1  = sc +  2L * N_NODES, *dn1 = sc + 3L * N_NODES;
  float* asm_= sc +  4L * N_NODES, *adm = sc + 5L * N_NODES;
  float* mm  = sc +  6L * N_NODES, *dnm = sc + 7L * N_NODES;
  float* asl = sc +  8L * N_NODES, *adl = sc + 9L * N_NODES;
  float* ml  = sc + 10L * N_NODES, *dnl = sc + 11L * N_NODES;
  float* eb1 = sc + 12L * N_NODES;
  float* ebm = eb1 + E_TOT;
  float* ebl = ebm + E_TOT;

  float* out_mu = (float*)d_out;
  float* out_ls = out_mu + SZ_O;

  const int TB = 256;
  const float NEG_INF = -__builtin_huge_valf();

  // ---- init accumulators (every call: graph replays must be deterministic)
  fill_f32<<<1024, TB, 0, stream>>>(h1, 0.0f, SZ_H);
  fill_f32<<<1024, TB, 0, stream>>>((float*)d_out, 0.0f, 2 * SZ_O);
  fill_f32<<<ceildiv(N_NODES, TB), TB, 0, stream>>>(m1, NEG_INF, N_NODES);
  fill_f32<<<ceildiv(N_NODES, TB), TB, 0, stream>>>(mm, NEG_INF, N_NODES);
  fill_f32<<<ceildiv(N_NODES, TB), TB, 0, stream>>>(ml, NEG_INF, N_NODES);
  fill_f32<<<ceildiv(N_NODES, TB), TB, 0, stream>>>(dn1, 0.0f, N_NODES);
  fill_f32<<<ceildiv(N_NODES, TB), TB, 0, stream>>>(dnm, 0.0f, N_NODES);
  fill_f32<<<ceildiv(N_NODES, TB), TB, 0, stream>>>(dnl, 0.0f, N_NODES);

  const int gemm_blocks = ceildiv(N_NODES / 16, 8);   // 1 wave = 16-row strip

  // ---- layer 1: hx = x @ W1 (WMMA + fused dots), softmax, aggregate, relu+b1
  wmma_gemm_attn<HID / 16, IN_CH><<<gemm_blocks, TB, 0, stream>>>(
      x, W1, hx, a_src1, a_dst1, as1, ad1, N_NODES);
  edge_logits_max<<<ceildiv(E_TOT, TB), TB, 0, stream>>>(ei, as1, ad1, eb1, m1,
                                                         nullptr, nullptr, nullptr, nullptr);
  edge_exp_sum<<<ceildiv(E_TOT, TB), TB, 0, stream>>>(ei, eb1, m1, dn1,
                                                      nullptr, nullptr, nullptr);
  edge_aggregate128<<<ceildiv(E_TOT, 8), TB, 0, stream>>>(ei, eb1, dn1, hx, h1);
  bias_act<<<ceildiv(SZ_H, TB), TB, 0, stream>>>(h1, b1, SZ_H, HID, 1);

  // ---- layers 2+3 (mu, logstd) fused over shared edges
  wmma_gemm_attn<OUT_CH / 16, HID><<<gemm_blocks, TB, 0, stream>>>(
      h1, W_mu, hmu, a_srcm, a_dstm, asm_, adm, N_NODES);
  wmma_gemm_attn<OUT_CH / 16, HID><<<gemm_blocks, TB, 0, stream>>>(
      h1, W_ls, hls, a_srcl, a_dstl, asl, adl, N_NODES);
  edge_logits_max<<<ceildiv(E_TOT, TB), TB, 0, stream>>>(ei, asm_, adm, ebm, mm,
                                                         asl, adl, ebl, ml);
  edge_exp_sum<<<ceildiv(E_TOT, TB), TB, 0, stream>>>(ei, ebm, mm, dnm, ebl, ml, dnl);
  edge_aggregate_dual64<<<ceildiv(E_TOT, 8), TB, 0, stream>>>(ei, ebm, dnm, hmu, out_mu,
                                                              ebl, dnl, hls, out_ls);
  bias_act<<<ceildiv(SZ_O, TB), TB, 0, stream>>>(out_mu, b_mu, SZ_O, OUT_CH, 0);
  bias_act<<<ceildiv(SZ_O, TB), TB, 0, stream>>>(out_ls, b_ls, SZ_O, OUT_CH, 0);
}
